// MultiHeadAttention_42880953483492
// MI455X (gfx1250) — matrix operand order repaired and checked by reading, not compiled, not run on hardware
//
#include <hip/hip_runtime.h>
#include <hip/hip_bf16.h>

#define BB_  2
#define SS_  2048
#define DD_  1024
#define HH_  16
#define DH_  64
#define MM_  (BB_*SS_)   // 4096 rows for the projection GEMMs

typedef __attribute__((ext_vector_type(16))) _Float16 v16h;
typedef __attribute__((ext_vector_type(8)))  float    v8f;

union FragU { v16h v; uint4 q[2]; };
union H8U   { uint4 q; _Float16 h[8]; };
union H4U   { uint2 u; _Float16 h[4]; };

static __device__ inline v8f vzero8() {
  v8f z = {0.f,0.f,0.f,0.f,0.f,0.f,0.f,0.f};
  return z;
}

static __device__ inline v8f wmma16(v16h a, v16h b, v8f c) {
  // v_wmma_f32_16x16x32_f16: (neg_a, A, neg_b, B, c_mod, C, reuse_a, reuse_b)
  return __builtin_amdgcn_wmma_f32_16x16x32_f16(false, a, false, b, (short)0, c, false, false);
}

// Async 16B global->LDS copy (GLOBAL_LOAD_ASYNC_TO_LDS_B128, ASYNCcnt-tracked).
// LDS aperture maps flat addr[31:0] -> LDS byte offset, so the low 32 bits of
// the generic shared pointer are the VDST LDS address.
static __device__ inline void async_ld16(void* lds_ptr, const void* gptr) {
  const unsigned off = (unsigned)(unsigned long long)lds_ptr;
  asm volatile("global_load_async_to_lds_b128 %0, %1, off"
               :: "v"(off), "v"(gptr)
               : "memory");
}
// Wait until at most 2 async ops outstanding (the just-issued next tile).
static __device__ inline void wait_async_le2() {
  asm volatile("s_wait_asynccnt 0x2" ::: "memory");
}
static __device__ inline void wait_async_le0() {
  asm volatile("s_wait_asynccnt 0x0" ::: "memory");
}

// A-matrix fragment (16x32 f16): lane<16 holds row M=lane, K runs {k0..k0+7, k0+16..k0+23};
// lanes 16-31 hold row M=lane-16, K runs shifted by +8.  (ISA 7.12.2 table)
static __device__ inline v16h load_frag_a(const _Float16* sm, int ld, int row_base, int k0, int lane) {
  const int hv = lane >> 4;
  const int r  = lane & 15;
  const _Float16* p = sm + (row_base + r) * ld + k0 + 8 * hv;
  FragU f;
  f.q[0] = *(const uint4*)(p);
  f.q[1] = *(const uint4*)(p + 16);
  return f.v;
}

// B-matrix fragment (32x16 f16), stored row-major as B^T rows (N x K): lane holds column N,
// K runs {k0+8*lanehalf..+7, k0+16+8*lanehalf..+7} (same K order as the A fragment).
static __device__ inline v16h load_frag_b(const _Float16* sm, int ld, int col_base, int k0, int lane) {
  const int hv = lane >> 4;
  const int n  = lane & 15;
  const _Float16* p = sm + (col_base + n) * ld + k0 + 8 * hv;
  FragU f;
  f.q[0] = *(const uint4*)(p);
  f.q[1] = *(const uint4*)(p + 16);
  return f.v;
}

// ---------------------------------------------------------------------------
// Kernel 1: fused QKV projection.  Y = X @ W^T + b, written head-split f16
// [B,H,S,DH].  blockIdx.z selects q/k/v.  Tiles: 128(M) x 128(N) x 32(K).
// fp32 sources need VGPR-path conversion, so these stay on global_load_b128.
// ---------------------------------------------------------------------------
__global__ __launch_bounds__(256) void qkv_proj_kernel(
    const float* __restrict__ xq, const float* __restrict__ xk, const float* __restrict__ xv,
    const float* __restrict__ Wq, const float* __restrict__ bq,
    const float* __restrict__ Wk, const float* __restrict__ bk,
    const float* __restrict__ Wv, const float* __restrict__ bv,
    _Float16* __restrict__ Qh, _Float16* __restrict__ Kh, _Float16* __restrict__ Vh)
{
  const int z = blockIdx.z;
  const float* X    = (z == 0) ? xq : (z == 1) ? xk : xv;
  const float* W    = (z == 0) ? Wq : (z == 1) ? Wk : Wv;
  const float* bias = (z == 0) ? bq : (z == 1) ? bk : bv;
  _Float16*   dst   = (z == 0) ? Qh : (z == 1) ? Kh : Vh;

  __shared__ alignas(16) _Float16 As[128][32];
  __shared__ alignas(16) _Float16 Bs[128][32];

  const int tid  = threadIdx.x;
  const int lane = tid & 31;
  const int w    = tid >> 5;
  const int wm   = w >> 2;        // 0..1  -> 64-row band
  const int wn   = w & 3;         // 0..3  -> 32-col band
  const int hv   = lane >> 4;
  const int l15  = lane & 15;
  const int m0   = blockIdx.y * 128;
  const int n0   = blockIdx.x * 128;

  v8f acc[4][2];
  #pragma unroll
  for (int i = 0; i < 4; ++i) { acc[i][0] = vzero8(); acc[i][1] = vzero8(); }

  for (int kt = 0; kt < DD_ / 32; ++kt) {
    // ---- stage fp32 tiles into LDS as f16 ----
    #pragma unroll
    for (int i = 0; i < 4; ++i) {
      const int idx = tid + i * 256;          // 0..1023
      const int row = idx >> 3;               // 0..127
      const int cg  = idx & 7;                // 8 groups of 4 floats
      {
        float4 f = *(const float4*)(X + (size_t)(m0 + row) * DD_ + kt * 32 + cg * 4);
        H4U t;
        t.h[0] = (_Float16)f.x; t.h[1] = (_Float16)f.y;
        t.h[2] = (_Float16)f.z; t.h[3] = (_Float16)f.w;
        *(uint2*)&As[row][cg * 4] = t.u;
      }
      {
        float4 f = *(const float4*)(W + (size_t)(n0 + row) * DD_ + kt * 32 + cg * 4);
        H4U t;
        t.h[0] = (_Float16)f.x; t.h[1] = (_Float16)f.y;
        t.h[2] = (_Float16)f.z; t.h[3] = (_Float16)f.w;
        *(uint2*)&Bs[row][cg * 4] = t.u;
      }
    }
    if (kt + 1 < DD_ / 32) {   // gfx1250 global_prefetch_b8 for next K tile
      __builtin_prefetch(X + (size_t)(m0 + (tid >> 1)) * DD_ + (kt + 1) * 32 + (tid & 1) * 16, 0, 3);
      __builtin_prefetch(W + (size_t)(n0 + (tid >> 1)) * DD_ + (kt + 1) * 32 + (tid & 1) * 16, 0, 3);
    }
    __syncthreads();

    v16h bf0 = load_frag_b(&Bs[0][0], 32, wn * 32 + 0,  0, lane);
    v16h bf1 = load_frag_b(&Bs[0][0], 32, wn * 32 + 16, 0, lane);
    #pragma unroll
    for (int ms = 0; ms < 4; ++ms) {
      v16h af = load_frag_a(&As[0][0], 32, wm * 64 + ms * 16, 0, lane);
      acc[ms][0] = wmma16(af, bf0, acc[ms][0]);
      acc[ms][1] = wmma16(af, bf1, acc[ms][1]);
    }
    __syncthreads();
  }

  // ---- epilogue: bias + head-split f16 store ----
  #pragma unroll
  for (int ms = 0; ms < 4; ++ms) {
    #pragma unroll
    for (int ns = 0; ns < 2; ++ns) {
      const int n = n0 + wn * 32 + ns * 16 + l15;
      const float bv2 = bias[n];
      const int hh = n >> 6;         // n / DH
      const int dd = n & 63;         // n % DH
      #pragma unroll
      for (int r = 0; r < 8; ++r) {
        const int m  = m0 + wm * 64 + ms * 16 + r + 8 * hv;
        const int bb = m >> 11;      // m / S
        const int ss = m & (SS_ - 1);
        dst[(((size_t)bb * HH_ + hh) * SS_ + ss) * DH_ + dd] =
            (_Float16)(acc[ms][ns][r] + bv2);
      }
    }
  }
}

// ---------------------------------------------------------------------------
// Kernel 2: flash-style attention per (b,h).  64 q-rows per block, 64-key
// tiles, online softmax in fp32, both GEMMs on v_wmma_f32_16x16x32_f16.
// K tiles: double-buffered async global->LDS DMA, one tile ahead; the DMA for
// tile t+1 overlaps all WMMA + softmax work of tile t (wait_asynccnt <= 2).
// ---------------------------------------------------------------------------
__global__ __launch_bounds__(256) void attn_kernel(
    const _Float16* __restrict__ Qh, const _Float16* __restrict__ Kh,
    const _Float16* __restrict__ Vh, _Float16* __restrict__ Ctx)
{
  __shared__ alignas(16) _Float16 qs[64 * 64];      // [q][d]
  __shared__ alignas(16) _Float16 ks[2][64 * 64];   // [key][d], double-buffered
  __shared__ alignas(16) _Float16 vts[64 * 64];     // transposed: [d][key]
  __shared__ alignas(16) float    sbuf[64 * 64];    // logits fp32
  __shared__ alignas(16) _Float16 pbuf[64 * 64];    // exp(P) f16
  __shared__ float rmax[64], rsum[64], cfac[64];
  __shared__ float red[64 * 4], red2[64 * 4];

  const int tid  = threadIdx.x;
  const int lane = tid & 31;
  const int w    = tid >> 5;
  const int m_sub = w & 3;            // q-subtile 0..3
  const int nsub0 = (w >> 2) * 2;     // n-subtiles {nsub0, nsub0+1}
  const int nsub1 = nsub0 + 1;
  const int hv  = lane >> 4;
  const int l15 = lane & 15;

  const int bh = blockIdx.y;
  const int bb = bh >> 4;             // bh / H
  const int hh = bh & 15;             // bh % H
  const int q0 = blockIdx.x * 64;
  const int NT = SS_ / 64;            // 32 key tiles

  const _Float16* Qbase = Qh + ((size_t)(bb * HH_ + hh) * SS_ + q0) * DH_;
  const _Float16* Kbase = Kh + (size_t)(bb * HH_ + hh) * SS_ * DH_;
  const _Float16* Vbase = Vh + (size_t)(bb * HH_ + hh) * SS_ * DH_;

  const int srow_ld = tid >> 3;       // staging row 0..31 (segment pass 0)
  const int sg_ld   = tid & 7;        // staging dh segment

  if (tid < 64) { rmax[tid] = -1.0e30f; rsum[tid] = 0.f; }
  // prologue: async Q tile + async K tile 0
  #pragma unroll
  for (int i = 0; i < 2; ++i) {
    const int sid = tid + i * 256;
    const int row = sid >> 3;
    const int sg  = sid & 7;
    async_ld16(&qs[row * 64 + sg * 8], Qbase + (size_t)row * DH_ + sg * 8);
  }
  #pragma unroll
  for (int i = 0; i < 2; ++i) {
    const int sid = tid + i * 256;
    const int row = sid >> 3;
    const int sg  = sid & 7;
    async_ld16(&ks[0][row * 64 + sg * 8], Kbase + (size_t)row * DH_ + sg * 8);
  }

  v8f o0 = vzero8(), o1 = vzero8();

  for (int kt = 0; kt < NT; ++kt) {
    const int cur = kt & 1;
    const int kbase = kt * 64;
    const _Float16* kcur = &ks[cur][0];

    // ---- stage NEXT K tile via async DMA (overlaps this tile's compute) ----
    if (kt + 1 < NT) {
      _Float16* knxt = &ks[cur ^ 1][0];
      #pragma unroll
      for (int i = 0; i < 2; ++i) {
        const int sid = tid + i * 256;
        const int row = sid >> 3;
        const int sg  = sid & 7;
        async_ld16(&knxt[row * 64 + sg * 8],
                   Kbase + (size_t)(kbase + 64 + row) * DH_ + sg * 8);
      }
      __builtin_prefetch(Vbase + (size_t)(kbase + 64 + (tid >> 2)) * DH_ + (tid & 3) * 16, 0, 3);
    }
    // ---- stage CURRENT V tile through VGPRs (transpose) ----
    #pragma unroll
    for (int i = 0; i < 2; ++i) {
      const int sid = tid + i * 256;
      const int row = sid >> 3;       // key index 0..63
      const int sg  = sid & 7;        // dh segment
      H8U u;
      u.q = *(const uint4*)(Vbase + (size_t)(kbase + row) * DH_ + sg * 8);
      #pragma unroll
      for (int j = 0; j < 8; ++j) vts[(sg * 8 + j) * 64 + row] = u.h[j];
    }
    if (kt + 1 < NT) wait_async_le2();   // current K (and Q on kt==0) complete
    else             wait_async_le0();
    __syncthreads();

    // ---- S = (Q @ K^T) * 1/sqrt(DH) ----
    v8f s0 = vzero8(), s1 = vzero8();
    #pragma unroll
    for (int kk = 0; kk < 64; kk += 32) {
      v16h af = load_frag_a(qs, 64, m_sub * 16, kk, lane);
      v16h b0 = load_frag_b(kcur, 64, nsub0 * 16, kk, lane);
      v16h b1 = load_frag_b(kcur, 64, nsub1 * 16, kk, lane);
      s0 = wmma16(af, b0, s0);
      s1 = wmma16(af, b1, s1);
    }
    #pragma unroll
    for (int r = 0; r < 8; ++r) {
      const int row = m_sub * 16 + r + 8 * hv;
      sbuf[row * 64 + nsub0 * 16 + l15] = s0[r] * 0.125f;
      sbuf[row * 64 + nsub1 * 16 + l15] = s1[r] * 0.125f;
    }
    __syncthreads();

    // ---- online softmax (4 threads per q-row) ----
    const int srow = tid >> 2;
    const int sq   = tid & 3;
    float mloc = -1.0e30f;
    #pragma unroll
    for (int j = 0; j < 16; ++j) mloc = fmaxf(mloc, sbuf[srow * 64 + sq * 16 + j]);
    red[srow * 4 + sq] = mloc;
    __syncthreads();
    float tmax = fmaxf(fmaxf(red[srow * 4 + 0], red[srow * 4 + 1]),
                       fmaxf(red[srow * 4 + 2], red[srow * 4 + 3]));
    const float omax = rmax[srow];
    const float nmax = fmaxf(omax, tmax);
    float psum = 0.f;
    #pragma unroll
    for (int j = 0; j < 16; ++j) {
      const float e = __expf(sbuf[srow * 64 + sq * 16 + j] - nmax);
      pbuf[srow * 64 + sq * 16 + j] = (_Float16)e;
      psum += e;
    }
    red2[srow * 4 + sq] = psum;
    __syncthreads();
    if (sq == 0) {
      const float c = __expf(omax - nmax);
      rsum[srow] = c * rsum[srow] + red2[srow * 4 + 0] + red2[srow * 4 + 1] +
                   red2[srow * 4 + 2] + red2[srow * 4 + 3];
      rmax[srow] = nmax;
      cfac[srow] = c;
    }
    __syncthreads();

    // ---- rescale O and accumulate P @ V ----
    #pragma unroll
    for (int r = 0; r < 8; ++r) {
      const float c = cfac[m_sub * 16 + r + 8 * hv];
      o0[r] *= c;
      o1[r] *= c;
    }
    #pragma unroll
    for (int kk = 0; kk < 64; kk += 32) {
      v16h af = load_frag_a(pbuf, 64, m_sub * 16, kk, lane);
      v16h b0 = load_frag_b(vts, 64, nsub0 * 16, kk, lane);
      v16h b1 = load_frag_b(vts, 64, nsub1 * 16, kk, lane);
      o0 = wmma16(af, b0, o0);
      o1 = wmma16(af, b1, o1);
    }
    __syncthreads();   // protect vts/pbuf (+ks[nxt] staging target) for next tile
  }

  // ---- normalize and store context (merged-head [B,S,D] layout, f16) ----
  (void)srow_ld; (void)sg_ld;
  #pragma unroll
  for (int r = 0; r < 8; ++r) {
    const int row = m_sub * 16 + r + 8 * hv;
    const float inv = 1.0f / rsum[row];
    const int sidx = q0 + row;
    _Float16* out = Ctx + (size_t)(bb * SS_ + sidx) * DD_ + hh * DH_;
    out[nsub0 * 16 + l15] = (_Float16)(o0[r] * inv);
    out[nsub1 * 16 + l15] = (_Float16)(o1[r] * inv);
  }
}

// ---------------------------------------------------------------------------
// Kernel 3: output projection.  out = Ctx(f16) @ Wo^T + bo, fp32 store.
// A tiles: double-buffered async global->LDS DMA one tile ahead.
// B tiles: fp32->f16 convert path, staged into the opposite buffer.
// ---------------------------------------------------------------------------
__global__ __launch_bounds__(256) void out_proj_kernel(
    const _Float16* __restrict__ Ctx, const float* __restrict__ Wo,
    const float* __restrict__ bo, float* __restrict__ out)
{
  __shared__ alignas(16) _Float16 As[2][128][32];
  __shared__ alignas(16) _Float16 Bs[2][128][32];

  const int tid  = threadIdx.x;
  const int lane = tid & 31;
  const int w    = tid >> 5;
  const int wm   = w >> 2;
  const int wn   = w & 3;
  const int hv   = lane >> 4;
  const int l15  = lane & 15;
  const int m0   = blockIdx.y * 128;
  const int n0   = blockIdx.x * 128;
  const int NT   = DD_ / 32;

  v8f acc[4][2];
  #pragma unroll
  for (int i = 0; i < 4; ++i) { acc[i][0] = vzero8(); acc[i][1] = vzero8(); }

  // prologue: stage tile 0 (A async, B convert)
  #pragma unroll
  for (int i = 0; i < 2; ++i) {
    const int sid = tid + i * 256;
    const int row = sid >> 2;
    const int sg  = sid & 3;
    async_ld16(&As[0][row][sg * 8], Ctx + (size_t)(m0 + row) * DD_ + sg * 8);
  }
  #pragma unroll
  for (int i = 0; i < 4; ++i) {
    const int idx = tid + i * 256;
    const int row = idx >> 3;
    const int cg  = idx & 7;
    float4 f = *(const float4*)(Wo + (size_t)(n0 + row) * DD_ + cg * 4);
    H4U t;
    t.h[0] = (_Float16)f.x; t.h[1] = (_Float16)f.y;
    t.h[2] = (_Float16)f.z; t.h[3] = (_Float16)f.w;
    *(uint2*)&Bs[0][row][cg * 4] = t.u;
  }

  for (int kt = 0; kt < NT; ++kt) {
    const int cur = kt & 1;
    // ---- stage NEXT tile into the other buffer ----
    if (kt + 1 < NT) {
      const int nxt = cur ^ 1;
      #pragma unroll
      for (int i = 0; i < 2; ++i) {
        const int sid = tid + i * 256;
        const int row = sid >> 2;
        const int sg  = sid & 3;
        async_ld16(&As[nxt][row][sg * 8],
                   Ctx + (size_t)(m0 + row) * DD_ + (kt + 1) * 32 + sg * 8);
      }
      #pragma unroll
      for (int i = 0; i < 4; ++i) {
        const int idx = tid + i * 256;
        const int row = idx >> 3;
        const int cg  = idx & 7;
        float4 f = *(const float4*)(Wo + (size_t)(n0 + row) * DD_ + (kt + 1) * 32 + cg * 4);
        H4U t;
        t.h[0] = (_Float16)f.x; t.h[1] = (_Float16)f.y;
        t.h[2] = (_Float16)f.z; t.h[3] = (_Float16)f.w;
        *(uint2*)&Bs[nxt][row][cg * 4] = t.u;
      }
      __builtin_prefetch(Wo + (size_t)(n0 + (tid >> 1)) * DD_ + (kt + 1) * 32 + (tid & 1) * 16, 0, 3);
      wait_async_le2();
    } else {
      wait_async_le0();
    }
    __syncthreads();   // publish tile kt (async A + ds-stored B)

    v16h bf0 = load_frag_b(&Bs[cur][0][0], 32, wn * 32 + 0,  0, lane);
    v16h bf1 = load_frag_b(&Bs[cur][0][0], 32, wn * 32 + 16, 0, lane);
    #pragma unroll
    for (int ms = 0; ms < 4; ++ms) {
      v16h af = load_frag_a(&As[cur][0][0], 32, wm * 64 + ms * 16, 0, lane);
      acc[ms][0] = wmma16(af, bf0, acc[ms][0]);
      acc[ms][1] = wmma16(af, bf1, acc[ms][1]);
    }
    __syncthreads();   // compute done before next staging overwrites buffers
  }

  #pragma unroll
  for (int ms = 0; ms < 4; ++ms) {
    #pragma unroll
    for (int ns = 0; ns < 2; ++ns) {
      const int n = n0 + wn * 32 + ns * 16 + l15;
      const float bv2 = bo[n];
      #pragma unroll
      for (int r = 0; r < 8; ++r) {
        const int m = m0 + wm * 64 + ms * 16 + r + 8 * hv;
        out[(size_t)m * DD_ + n] = acc[ms][ns][r] + bv2;
      }
    }
  }
}

// ---------------------------------------------------------------------------
extern "C" void kernel_launch(void* const* d_in, const int* in_sizes, int n_in,
                              void* d_out, int out_size, void* d_ws, size_t ws_size,
                              hipStream_t stream) {
  (void)in_sizes; (void)n_in; (void)out_size; (void)ws_size;
  const float* q  = (const float*)d_in[0];
  const float* k  = (const float*)d_in[1];
  const float* v  = (const float*)d_in[2];
  const float* Wq = (const float*)d_in[3];
  const float* bq = (const float*)d_in[4];
  const float* Wk = (const float*)d_in[5];
  const float* bk = (const float*)d_in[6];
  const float* Wv = (const float*)d_in[7];
  const float* bv = (const float*)d_in[8];
  const float* Wo = (const float*)d_in[9];
  const float* bo = (const float*)d_in[10];
  float* out = (float*)d_out;

  const size_t NE = (size_t)MM_ * DD_;      // 4,194,304 elements per tensor
  _Float16* wsh = (_Float16*)d_ws;          // 4 x 8 MB = 32 MB of f16 scratch
  _Float16* Qh  = wsh;
  _Float16* Kh  = wsh + NE;
  _Float16* Vh  = wsh + 2 * NE;
  _Float16* Ctx = wsh + 3 * NE;

  qkv_proj_kernel<<<dim3(DD_ / 128, MM_ / 128, 3), 256, 0, stream>>>(
      q, k, v, Wq, bq, Wk, bk, Wv, bv, Qh, Kh, Vh);

  attn_kernel<<<dim3(SS_ / 64, BB_ * HH_), 256, 0, stream>>>(Qh, Kh, Vh, Ctx);

  out_proj_kernel<<<dim3(DD_ / 128, MM_ / 128), 256, 0, stream>>>(Ctx, Wo, bo, out);
}
